// RationaleSelectorModel_26268019982547
// MI455X (gfx1250) — matrix operand order repaired and checked by reading, not compiled, not run on hardware
//
#include <hip/hip_runtime.h>
#include <stdint.h>

// ---------------------------------------------------------------------------
// Nearest-centroid gating, fused GEMM+argmin for MI455X (gfx1250, wave32).
//
//   score_k = 0.5*||c_k||^2 - x . c_k          (x^2 term dropped: argmin-invariant)
//   out = gates * attn * (argmin_k score_k == entity_cluster)
//
// Compute: error-compensated bf16 split (x=x_hi+x_lo, c=c_hi+c_lo),
//   x.c ~= x_hi.c_hi + x_hi.c_lo + x_lo.c_hi   -> 3x v_wmma_f32_16x16x32_bf16
// hi/lo are SEPARATE planes (LDS + prepacked centroid planes in global) laid
// out in WMMA fragment half-order, so fragment builds are pure ds_load_b128.
// Centroid staging uses gfx1250 GLOBAL_LOAD_ASYNC_TO_LDS_B128 (ASYNCcnt),
// B fragments are software-pipelined one subtile ahead of the WMMA triples.
// ---------------------------------------------------------------------------

typedef __attribute__((ext_vector_type(16))) __bf16        v16bf;
typedef __attribute__((ext_vector_type(8)))  float         v8f;
typedef __attribute__((ext_vector_type(8)))  unsigned int  v8u;

constexpr int D      = 768;
constexpr int K      = 1024;
constexpr int NTOK   = 64 * 1024;   // B*L
constexpr int M_TILE = 128;         // tokens per workgroup (8 waves x 16)
constexpr int N_TILE = 128;         // centroids resident in LDS per pass
constexpr int D_STEP = 32;          // WMMA K per step (16 dwords of bf16 pairs)
constexpr int PROW   = 20;          // LDS plane row stride (dwords): mult-of-4
                                    // (b128 aligned); 20r mod 64 covers all
                                    // 4-aligned bank starts -> conflict-free

// hi = bf16 truncation of x, lo = bf16 truncation of (x - hi).
__device__ __forceinline__ void split2(float x, unsigned& hi16, unsigned& lo16) {
  unsigned u  = __builtin_bit_cast(unsigned, x);
  unsigned hm = u & 0xFFFF0000u;
  float    lf = x - __builtin_bit_cast(float, hm);
  hi16 = u >> 16;
  lo16 = __builtin_bit_cast(unsigned, lf) >> 16;
}

// Pack element pair (even k, odd k) into one plane dword: even in [15:0].
__device__ __forceinline__ void packPair(float e0, float e1,
                                         unsigned& hiD, unsigned& loD) {
  unsigned h0, l0, h1, l1;
  split2(e0, h0, l0);
  split2(e1, h1, l1);
  hiD = h0 | (h1 << 16);
  loD = l0 | (l1 << 16);
}

// Centroid prep: emit separate hi/lo half-planes (dword = bf16 pair along K).
__global__ void pack_centroid_planes_kernel(const float* __restrict__ c,
                                            unsigned* __restrict__ cHi,
                                            unsigned* __restrict__ cLo, int npair) {
  int i = blockIdx.x * blockDim.x + threadIdx.x;
  if (i < npair) {
    unsigned hiD, loD;
    packPair(c[2 * i], c[2 * i + 1], hiD, loD);
    cHi[i] = hiD;
    cLo[i] = loD;
  }
}

__global__ void centroid_halfnorm_kernel(const float* __restrict__ c,
                                         float* __restrict__ hc2) {
  int k = blockIdx.x * blockDim.x + threadIdx.x;
  if (k < K) {
    const float* row = c + (size_t)k * D;
    float s = 0.f;
    for (int d = 0; d < D; ++d) { float x = row[d]; s = fmaf(x, x, s); }
    hc2[k] = 0.5f * s;
  }
}

// gfx1250 async global->LDS copy, 16 bytes per lane. Tracked by ASYNCcnt.
// vdst VGPR carries the LDS byte offset; vaddr is the 64-bit global address.
__device__ __forceinline__ void asyncCopyB128(unsigned ldsByteOff,
                                              const unsigned* gsrc) {
  asm volatile("global_load_async_to_lds_b128 %0, %1, off"
               :: "v"(ldsByteOff), "v"(gsrc)
               : "memory");
}
__device__ __forceinline__ void waitAsync0() {
  asm volatile("s_wait_asynccnt 0" ::: "memory");
}

// Assemble a 16-half fragment from two 16-byte LDS runs.
__device__ __forceinline__ v16bf fragFrom(const unsigned* p0, const unsigned* p1) {
  uint4 x = *reinterpret_cast<const uint4*>(p0);
  uint4 y = *reinterpret_cast<const uint4*>(p1);
  v8u v;
  v[0] = x.x; v[1] = x.y; v[2] = x.z; v[3] = x.w;
  v[4] = y.x; v[5] = y.y; v[6] = y.z; v[7] = y.w;
  return __builtin_bit_cast(v16bf, v);
}

__global__ __launch_bounds__(256)
void cluster_gate_kernel(const float* __restrict__ tok,
                         const unsigned* __restrict__ cHi,
                         const unsigned* __restrict__ cLo,
                         const float* __restrict__ hc2,
                         const float* __restrict__ gates,
                         const float* __restrict__ attn,
                         const int* __restrict__ ecp,
                         float* __restrict__ out) {
  __shared__ __align__(16) unsigned sAhi[M_TILE * PROW];
  __shared__ __align__(16) unsigned sAlo[M_TILE * PROW];
  __shared__ __align__(16) unsigned sChi[N_TILE * PROW];
  __shared__ __align__(16) unsigned sClo[N_TILE * PROW];

  const int t      = threadIdx.x;
  const int lane   = t & 31;
  const int wave   = t >> 5;         // 8 waves
  const int h      = lane >> 4;      // lane half (0/1)
  const int r      = lane & 15;
  const int wgBase = blockIdx.x * M_TILE;

  const int srow   = t >> 1;                 // staging: 2 threads per row
  const int sdoff  = (t & 1) * 8;            // dword offset within row
  const int sbase  = srow * PROW + sdoff;

  // Per-thread fixed LDS byte offsets for async centroid staging.
  const unsigned ldsChi0 = (unsigned)(uintptr_t)&sChi[sbase];
  const unsigned ldsChi1 = (unsigned)(uintptr_t)&sChi[sbase + 4];
  const unsigned ldsClo0 = (unsigned)(uintptr_t)&sClo[sbase];
  const unsigned ldsClo1 = (unsigned)(uintptr_t)&sClo[sbase + 4];

  float bval[8];
  int   bidx[8];
#pragma unroll
  for (int i = 0; i < 8; ++i) { bval[i] = 3.4e38f; bidx[i] = 0; }

  const v8f vzero = {0.f, 0.f, 0.f, 0.f, 0.f, 0.f, 0.f, 0.f};

  for (int nt = 0; nt < K / N_TILE; ++nt) {
    const int centBase = nt * N_TILE;
    v8f acc[8];
#pragma unroll
    for (int i = 0; i < 8; ++i) acc[i] = vzero;

    for (int ds = 0; ds < D / D_STEP; ++ds) {
      const int d0 = ds * D_STEP;
      __syncthreads();  // previous iteration's LDS reads complete
      {
        // ---- stage C asynchronously (pure copy of pre-split planes) ----
        const size_t cOff = (size_t)(centBase + srow) * (D / 2) + d0 / 2 + sdoff;
        asyncCopyB128(ldsChi0, cHi + cOff);
        asyncCopyB128(ldsChi1, cHi + cOff + 4);
        asyncCopyB128(ldsClo0, cLo + cOff);
        asyncCopyB128(ldsClo1, cLo + cOff + 4);

        // ---- stage A: token f32 -> hi/lo planes (needs split conversion) ----
        const float4* g4 = reinterpret_cast<const float4*>(
            tok + (size_t)(wgBase + srow) * D + d0 + (t & 1) * 16);
        unsigned hiD[8], loD[8];
#pragma unroll
        for (int i4 = 0; i4 < 4; ++i4) {
          float4 v = g4[i4];
          packPair(v.x, v.y, hiD[2 * i4 + 0], loD[2 * i4 + 0]);
          packPair(v.z, v.w, hiD[2 * i4 + 1], loD[2 * i4 + 1]);
        }
        *reinterpret_cast<uint4*>(&sAhi[sbase + 0]) = make_uint4(hiD[0], hiD[1], hiD[2], hiD[3]);
        *reinterpret_cast<uint4*>(&sAhi[sbase + 4]) = make_uint4(hiD[4], hiD[5], hiD[6], hiD[7]);
        *reinterpret_cast<uint4*>(&sAlo[sbase + 0]) = make_uint4(loD[0], loD[1], loD[2], loD[3]);
        *reinterpret_cast<uint4*>(&sAlo[sbase + 4]) = make_uint4(loD[4], loD[5], loD[6], loD[7]);
      }
      waitAsync0();     // async centroid copies landed in LDS
      __syncthreads();  // (also waits the A-plane ds_stores)

      // ---- A fragments: dword j covers K pair 2j+16h ->
      //      plane dwords {8h..8h+7}: 2x b128/plane ----
      const int abase = (wave * 16 + r) * PROW + 8 * h;
      const v16bf a_hi = fragFrom(&sAhi[abase], &sAhi[abase + 4]);
      const v16bf a_lo = fragFrom(&sAlo[abase], &sAlo[abase + 4]);

      // ---- B fragments: dword j covers K pair 2j+16h -> run {8h..8h+7};
      //      software-pipelined one subtile ahead of the WMMA triples ----
      const int bbase0 = r * PROW + 8 * h;
      v16bf b_hi = fragFrom(&sChi[bbase0], &sChi[bbase0 + 4]);
      v16bf b_lo = fragFrom(&sClo[bbase0], &sClo[bbase0 + 4]);

#pragma unroll
      for (int ns = 0; ns < 8; ++ns) {
        v16bf nb_hi, nb_lo;
        if (ns < 7) {
          const int bbase = ((ns + 1) * 16 + r) * PROW + 8 * h;
          nb_hi = fragFrom(&sChi[bbase], &sChi[bbase + 4]);
          nb_lo = fragFrom(&sClo[bbase], &sClo[bbase + 4]);
        }
        acc[ns] = __builtin_amdgcn_wmma_f32_16x16x32_bf16(
            false, a_hi, false, b_hi, (short)0, acc[ns], false, false);
        acc[ns] = __builtin_amdgcn_wmma_f32_16x16x32_bf16(
            false, a_hi, false, b_lo, (short)0, acc[ns], false, false);
        acc[ns] = __builtin_amdgcn_wmma_f32_16x16x32_bf16(
            false, a_lo, false, b_hi, (short)0, acc[ns], false, false);
        b_hi = nb_hi;
        b_lo = nb_lo;
      }
    }

    // ---- fused score + running argmin (centroid ids monotonic -> strict <
    //      reproduces jnp.argmin first-min semantics) ----
#pragma unroll
    for (int ns = 0; ns < 8; ++ns) {
      const int   kidx = centBase + ns * 16 + r;
      const float c2h  = hc2[kidx];
#pragma unroll
      for (int rr = 0; rr < 8; ++rr) {
        float s = c2h - acc[ns][rr];
        if (s < bval[rr]) { bval[rr] = s; bidx[rr] = kidx; }
      }
    }
  }

  // ---- cross-lane argmin within each 16-lane half (C slot rr <-> M=rr+8h) ----
#pragma unroll
  for (int off = 1; off < 16; off <<= 1) {
#pragma unroll
    for (int rr = 0; rr < 8; ++rr) {
      float ov = __shfl_xor(bval[rr], off, 32);
      int   oi = __shfl_xor(bidx[rr], off, 32);
      if (ov < bval[rr] || (ov == bval[rr] && oi < bidx[rr])) {
        bval[rr] = ov; bidx[rr] = oi;
      }
    }
  }

  if (r == 0) {  // lanes 0 and 16 write 8 tokens each
    const int ec = *ecp;
#pragma unroll
    for (int rr = 0; rr < 8; ++rr) {
      const int   ti = wgBase + wave * 16 + 8 * h + rr;
      const float em = (bidx[rr] == ec) ? 1.0f : 0.0f;
      out[ti] = gates[ti] * attn[ti] * em;
    }
  }
}

// ---- scalar fallback (only if workspace is too small for centroid planes) ----
__global__ void cluster_gate_fallback(const float* __restrict__ tok,
                                      const float* __restrict__ cents,
                                      const float* __restrict__ gates,
                                      const float* __restrict__ attn,
                                      const int* __restrict__ ecp,
                                      float* __restrict__ out) {
  __shared__ float sx[D];
  __shared__ float sv[256];
  __shared__ int   si[256];
  const int ti = blockIdx.x;
  const int t  = threadIdx.x;
  for (int d = t; d < D; d += 256) sx[d] = tok[(size_t)ti * D + d];
  __syncthreads();
  float bv = 3.4e38f; int bi = 0;
  for (int k = t; k < K; k += 256) {
    const float* c = cents + (size_t)k * D;
    float dot = 0.f, c2 = 0.f;
    for (int d = 0; d < D; ++d) { float cv = c[d]; dot = fmaf(sx[d], cv, dot); c2 = fmaf(cv, cv, c2); }
    float s = 0.5f * c2 - dot;
    if (s < bv) { bv = s; bi = k; }
  }
  sv[t] = bv; si[t] = bi;
  __syncthreads();
  for (int off = 128; off > 0; off >>= 1) {
    if (t < off) {
      if (sv[t + off] < sv[t] || (sv[t + off] == sv[t] && si[t + off] < si[t])) {
        sv[t] = sv[t + off]; si[t] = si[t + off];
      }
    }
    __syncthreads();
  }
  if (t == 0) {
    float em = (si[0] == *ecp) ? 1.0f : 0.0f;
    out[ti] = gates[ti] * attn[ti] * em;
  }
}

extern "C" void kernel_launch(void* const* d_in, const int* in_sizes, int n_in,
                              void* d_out, int out_size, void* d_ws, size_t ws_size,
                              hipStream_t stream) {
  const float* tok   = (const float*)d_in[0];  // [B*L, D] f32
  const float* gates = (const float*)d_in[1];  // [B*L]
  const float* attn  = (const float*)d_in[2];  // [B*L]
  const float* cents = (const float*)d_in[3];  // [K, D]
  const int*   ecp   = (const int*)d_in[4];    // scalar
  float*       outp  = (float*)d_out;

  const size_t planeDw = (size_t)K * D / 2;    // dwords per half-plane
  const size_t needWs  = planeDw * 2 * sizeof(unsigned) + (size_t)K * sizeof(float);
  if (ws_size < needWs) {
    cluster_gate_fallback<<<NTOK, 256, 0, stream>>>(tok, cents, gates, attn, ecp, outp);
    return;
  }

  unsigned* cHi = (unsigned*)d_ws;
  unsigned* cLo = cHi + planeDw;
  float*    hc2 = (float*)(cLo + planeDw);

  pack_centroid_planes_kernel<<<((int)planeDw + 255) / 256, 256, 0, stream>>>(
      cents, cHi, cLo, (int)planeDw);
  centroid_halfnorm_kernel<<<(K + 255) / 256, 256, 0, stream>>>(cents, hc2);
  cluster_gate_kernel<<<NTOK / M_TILE, 256, 0, stream>>>(
      tok, cHi, cLo, hc2, gates, attn, ecp, outp);
}
